// DepthConv_73564199846424
// MI455X (gfx1250) — matrix operand order repaired and checked by reading, not compiled, not run on hardware
//
#include <hip/hip_runtime.h>

// ---------------------------------------------------------------------------
// Depth-weighted 3x3 conv, per-tap GEMM formulation (bf16 WMMA, f32 acc):
//   out[o,p] = bias[o] + sum_tap dw[tap,p] * ( sum_c weight[o,c,tap] * img_pad[c, p@tap] )
// dw is channel-independent -> scale the per-tap 16x16 GEMM tile (lane = pixel,
// so dw is a per-lane scalar). 64-pixel strips: each wave runs 2 pixel sub-tiles
// with the same A fragments in registers (A-load amortization 2x).
// ---------------------------------------------------------------------------

typedef __attribute__((ext_vector_type(16))) __bf16          v16bf;
typedef __attribute__((ext_vector_type(16))) unsigned short  v16u;
typedef __attribute__((ext_vector_type(8)))  float           v8f;

#define HW_   128
#define CIN_  64
#define COUT_ 64
#define NCOL  66        // 64-pixel strip + 2 halo columns
#define CPAD  80        // channel dim padded 64->80 ushorts (160B: 32B-aligned, 2-way banks)

__device__ __forceinline__ unsigned short f2bf(float x) {
    unsigned int u = __float_as_uint(x);
    unsigned int r = u + 0x7FFFu + ((u >> 16) & 1u);   // round-to-nearest-even
    return (unsigned short)(r >> 16);
}

// ---------------------------------------------------------------------------
// Prep: fp32 weight [64,64,3,3] -> bf16 A fragments in WMMA per-lane order.
// Fragment index = ((mt*9 + tap)*2 + s2)*32 + lane ; element e of A(16x32 bf16):
//   M = mt*16 + (lane & 15)
//   C = s2*32 + e + ((lane >= 16) ? 16 : 0)
// ---------------------------------------------------------------------------
__global__ __launch_bounds__(256)
void prep_weights_kernel(const float* __restrict__ weight,
                         unsigned short* __restrict__ wA)
{
    int g = blockIdx.x * 256 + threadIdx.x;
    if (g >= 4 * 9 * 2 * 32 * 16) return;
    int e    = g & 15;
    int t    = g >> 4;
    int lane = t & 31;  t >>= 5;
    int s2   = t & 1;   t >>= 1;
    int tap  = t % 9;
    int mt   = t / 9;

    int c = (s2 << 5) + e + ((lane >> 4) << 4);
    int m = (mt << 4) + (lane & 15);
    wA[g] = f2bf(weight[(m * CIN_ + c) * 9 + tap]);
}

// ---------------------------------------------------------------------------
// Main: one workgroup = (b, h, 64-wide pixel strip). 8 wave32:
// 4 M-tiles x 2 N-halves, each wave covers 2 x 16-pixel sub-tiles.
// ---------------------------------------------------------------------------
__global__ __launch_bounds__(256)
void depthconv_wmma_kernel(const float* __restrict__ img,
                           const float* __restrict__ depth,
                           const float* __restrict__ bias,
                           const unsigned short* __restrict__ wA,
                           float* __restrict__ out)
{
    __shared__ alignas(32) unsigned short sImg[3 * NCOL * CPAD]; // [row][col][ch] bf16, 31680 B
    __shared__ float sDW[9 * 64];                                // [tap][pixel]

    const int tid = threadIdx.x;
    const int blk = blockIdx.x;
    const int w0  = (blk & 1) * 64;
    const int h   = (blk >> 1) & (HW_ - 1);
    const int b   = blk >> 8;

    // ---- depth weights dw[tap][p] (shared by all 4 M-tile waves) -------
    const float* depth_b = depth + b * (HW_ * HW_);
    for (int idx = tid; idx < 9 * 64; idx += 256) {
        int tap = idx >> 6, p = idx & 63;
        int i = tap / 3, j = tap % 3;
        int w = w0 + p;
        float center = depth_b[h * HW_ + w];
        int r = h + i - 1, cc = w + j - 1;
        float dp = (r >= 0 && r < HW_ && cc >= 0 && cc < HW_)
                     ? depth_b[r * HW_ + cc] : 0.0f;
        sDW[idx] = __expf(-8.3f * fabsf(center - dp));
    }

    // ---- stage raw bf16 img window [3][66][64ch] (zero halo) -----------
    // item = (row, cgroup, col): lanes sweep col -> coalesced global loads,
    // one contiguous 32B LDS vector store per item.
    const float* img_b = img + b * (CIN_ * HW_ * HW_);
    for (int f = tid; f < 3 * 4 * NCOL; f += 256) {
        int col = f % NCOL;
        int t   = f / NCOL;
        int cg  = t & 3;      // channel group of 16
        int row = t >> 2;
        int r  = h + row - 1;
        int cc = w0 + col - 1;
        bool inb = (r >= 0) && (r < HW_) && (cc >= 0) && (cc < HW_);
        v16u vals;
#pragma unroll
        for (int e = 0; e < 16; ++e) {
            int c = (cg << 4) + e;
            vals[e] = inb ? f2bf(img_b[(c * HW_ + r) * HW_ + cc]) : (unsigned short)0;
        }
        *reinterpret_cast<v16u*>(&sImg[(row * NCOL + col) * CPAD + (cg << 4)]) = vals;
    }
    __syncthreads();

    // ---- per-tap WMMA GEMM ---------------------------------------------
    const int wave = tid >> 5, lane = tid & 31;
    const int mt = wave & 3;                       // Cout tile
    const int nt = wave >> 2;                      // pixel half (32 px)
    const int p0    = (nt << 5) + (lane & 15);     // first sub-tile pixel [0,64)
    const int khalf = (lane & 16);                 // B frag: lane>=16 holds C+16..C+31

    v8f accs[2] = { v8f{}, v8f{} };
    const v16bf* A = reinterpret_cast<const v16bf*>(wA) + (mt * 18) * 32 + lane;

#pragma unroll
    for (int tap = 0; tap < 9; ++tap) {
        const int row = tap / 3, j = tap % 3;
        v16bf a0 = A[(tap * 2 + 0) * 32];          // loaded once, used by both sub-tiles
        v16bf a1 = A[(tap * 2 + 1) * 32];
        const unsigned short* rbase = &sImg[(row * NCOL + j) * CPAD + khalf];
#pragma unroll
        for (int sub = 0; sub < 2; ++sub) {
            const int pl = p0 + (sub << 4);
            const unsigned short* bp = rbase + pl * CPAD;
            v16bf b0 = *reinterpret_cast<const v16bf*>(bp);        // c in [0,32)
            v16bf b1 = *reinterpret_cast<const v16bf*>(bp + 32);   // c in [32,64)
            v8f tacc = {};
            tacc = __builtin_amdgcn_wmma_f32_16x16x32_bf16(false, a0, false, b0,
                                                           (short)0, tacc, false, false);
            tacc = __builtin_amdgcn_wmma_f32_16x16x32_bf16(false, a1, false, b1,
                                                           (short)0, tacc, false, false);
            float dwv = sDW[tap * 64 + pl];
#pragma unroll
            for (int r = 0; r < 8; ++r)
                accs[sub][r] = fmaf(tacc[r], dwv, accs[sub][r]);
        }
    }

    // ---- store D: lane = pixel column; VGPR r -> m = r + 8*(lane>=16) --
    const int mbase = (mt << 4) + ((lane >> 4) << 3);
#pragma unroll
    for (int sub = 0; sub < 2; ++sub) {
        const int w = w0 + p0 + (sub << 4);
#pragma unroll
        for (int r = 0; r < 8; ++r) {
            int o = mbase + r;
            out[((b * COUT_ + o) * HW_ + h) * HW_ + w] = accs[sub][r] + bias[o];
        }
    }
}

// ---------------------------------------------------------------------------
extern "C" void kernel_launch(void* const* d_in, const int* in_sizes, int n_in,
                              void* d_out, int out_size, void* d_ws, size_t ws_size,
                              hipStream_t stream)
{
    const float* img    = (const float*)d_in[0];  // [8,64,128,128]
    const float* depth  = (const float*)d_in[1];  // [8,1,128,128]
    const float* weight = (const float*)d_in[2];  // [64,64,3,3]
    const float* bias   = (const float*)d_in[3];  // [1,64]
    float*       outp   = (float*)d_out;          // [8,64,128,128]
    unsigned short* wA  = (unsigned short*)d_ws;  // 73728 B of A fragments

    // 4*9*2*32*16 = 36864 fragment elements -> 144 blocks of 256
    prep_weights_kernel<<<144, 256, 0, stream>>>(weight, wA);

    // 8 (B) * 128 (H) * 2 (W/64) = 2048 workgroups
    depthconv_wmma_kernel<<<2048, 256, 0, stream>>>(img, depth, bias, wA, outp);
}
